// GNN_996432413617
// MI455X (gfx1250) — matrix operand order repaired and checked by reading, not compiled, not run on hardware
//
#include <hip/hip_runtime.h>

// Problem constants (match reference)
#define NN 50000          // nodes
#define NE 800000         // edges (directed edges = 2*NE)
#define NG 64             // graphs

typedef __attribute__((ext_vector_type(16))) _Float16 v16h;
typedef __attribute__((ext_vector_type(4)))  _Float16 v4h;
typedef __attribute__((ext_vector_type(8)))  float    v8f;

// ---------------------------------------------------------------------------
// WMMA helpers (CDNA5 wave32, V_WMMA_F32_16X16X32_F16)
// ---------------------------------------------------------------------------
__device__ __forceinline__ v8f wmma_f16(v16h a, v16h b, v8f c) {
  return __builtin_amdgcn_wmma_f32_16x16x32_f16(false, a, false, b, (short)0, c,
                                                false, false);
}

// A fragment: 16 rows x 32-K chunk from an LDS row-major tile [16][ld] (f16).
// ISA 7.12.2 16-bit A 16x32 layout. K-contiguous per half-run -> compiler
// merges into 2x ds_load_b128 per fragment.
__device__ __forceinline__ v16h fragA(const _Float16* s, int k0, int ld, int lane) {
  const int m  = lane & 15;
  const int hi = lane >> 4;
  v16h a;
#pragma unroll
  for (int h = 0; h < 16; ++h) {
    int k = k0 + h + ((h & 8) ? 8 : 0) + (hi ? 8 : 0);
    a[h] = s[m * ld + k];
  }
  return a;
}

// B fragment from a fragment-ordered packed weight buffer:
//   packed[((frag)*32 + lane)*16 + h]  ->  one 32-byte vector load per lane.
__device__ __forceinline__ v16h fragBp(const _Float16* __restrict__ wp, int f,
                                       int lane) {
  return ((const v16h*)wp)[f * 32 + lane];
}

// Broadcast bias into a C accumulator (free vs zero-init; kills epilogue adds)
__device__ __forceinline__ v8f biasC(float bb) {
  v8f c;
#pragma unroll
  for (int r = 0; r < 8; ++r) c[r] = bb;
  return c;
}

// ReLU applied on the f16 side of the convert: one v_cvt + one 16-bit max
// (identical result to f32-side ReLU; avoids LLVM's canonicalize+max pair).
__device__ __forceinline__ _Float16 relu16(float x) {
  return __builtin_fmaxf16((_Float16)x, (_Float16)0.f);
}

// ---------------------------------------------------------------------------
// Weight repack: f32 row-major [K][Nt] -> f16 fragment order.
// Fragment f = kc*(Nt/16) + jn holds the 32x16 tile (K chunk kc, N tile jn).
// Lane layout (16-bit B): element h of lane half hi holds K=k0+8*hi+h (h<8),
// K=k0+16+8*hi+(h-8) (h>=8); N = lane%16; 16 halves per lane stored contiguously.
// ---------------------------------------------------------------------------
__global__ void repackB_kernel(const float* __restrict__ w,
                               _Float16* __restrict__ wp, int K, int Nt) {
  int idx = blockIdx.x * blockDim.x + threadIdx.x;
  if (idx >= K * Nt) return;
  int h    = idx & 15;
  int lane = (idx >> 4) & 31;
  int f    = idx >> 9;
  int JN   = Nt >> 4;
  int kc   = f / JN, jn = f % JN;
  int k = kc * 32 + h + ((h & 8) ? 8 : 0) + ((lane >> 4) ? 8 : 0);
  int n = jn * 16 + (lane & 15);
  wp[idx] = (_Float16)w[k * Nt + n];
}

// ---------------------------------------------------------------------------
// Node combiner: nf = relu(relu(x @ wn1 + bn1) @ wn2 + bn2)   [NN,64] f16 out
// One 16-row tile per wave (no stride loop -> low register residency).
// ---------------------------------------------------------------------------
__global__ void __launch_bounds__(256)
node_combiner_kernel(const float* __restrict__ nf,       // [NN,64] f32
                     const _Float16* __restrict__ w1p,   // packed [64,128]
                     const float* __restrict__ b1,       // [128]
                     const _Float16* __restrict__ w2p,   // packed [128,64]
                     const float* __restrict__ b2,       // [64]
                     _Float16* __restrict__ outh) {      // [NN,64] f16
  __shared__ __align__(16) _Float16 sm[8][16 * 64 + 16 * 128];
  const int wave = threadIdx.x >> 5;
  const int lane = threadIdx.x & 31;
  _Float16* sA = sm[wave];            // 16x64 input tile
  _Float16* sH = sm[wave] + 16 * 64;  // 16x128 hidden tile
  const int n = lane & 15, hi = lane >> 4;
  const int t = blockIdx.x * 8 + wave;
  if (t >= NN / 16) return;

#pragma unroll
  for (int r = 0; r < 16; ++r) {
    if (lane < 16) {
      float4 v = ((const float4*)(nf + (size_t)(t * 16 + r) * 64))[lane];
      v4h h;
      h[0] = (_Float16)v.x; h[1] = (_Float16)v.y;
      h[2] = (_Float16)v.z; h[3] = (_Float16)v.w;
      ((v4h*)(sA + r * 64))[lane] = h;
    }
  }
  v16h a0 = fragA(sA, 0, 64, lane);
  v16h a1 = fragA(sA, 32, 64, lane);
  // GEMM1: 64 -> 128 (bias in C), relu, restage
#pragma unroll
  for (int j = 0; j < 8; ++j) {
    v8f c = biasC(b1[j * 16 + n]);
    c = wmma_f16(a0, fragBp(w1p, 0 * 8 + j, lane), c);
    c = wmma_f16(a1, fragBp(w1p, 1 * 8 + j, lane), c);
#pragma unroll
    for (int r = 0; r < 8; ++r)
      sH[(r + 8 * hi) * 128 + j * 16 + n] = relu16(c[r]);
  }
  v16h h0 = fragA(sH, 0, 128, lane);
  v16h h1 = fragA(sH, 32, 128, lane);
  v16h h2 = fragA(sH, 64, 128, lane);
  v16h h3 = fragA(sH, 96, 128, lane);
  // GEMM2: 128 -> 64, relu, store f16
#pragma unroll
  for (int j = 0; j < 4; ++j) {
    v8f c = biasC(b2[j * 16 + n]);
    c = wmma_f16(h0, fragBp(w2p, 0 * 4 + j, lane), c);
    c = wmma_f16(h1, fragBp(w2p, 1 * 4 + j, lane), c);
    c = wmma_f16(h2, fragBp(w2p, 2 * 4 + j, lane), c);
    c = wmma_f16(h3, fragBp(w2p, 3 * 4 + j, lane), c);
#pragma unroll
    for (int r = 0; r < 8; ++r)
      outh[(size_t)(t * 16 + r + 8 * hi) * 64 + j * 16 + n] = relu16(c[r]);
  }
}

// ---------------------------------------------------------------------------
// Edge combiner: ef = relu(relu(e @ we1 + be1) @ we2 + be2)   [NE,32] f16 out
// One 16-row tile per wave.
// ---------------------------------------------------------------------------
__global__ void __launch_bounds__(256)
edge_combiner_kernel(const float* __restrict__ ef,      // [NE,32] f32
                     const _Float16* __restrict__ w1p,  // packed [32,64]
                     const float* __restrict__ b1,      // [64]
                     const _Float16* __restrict__ w2p,  // packed [64,32]
                     const float* __restrict__ b2,      // [32]
                     _Float16* __restrict__ outh) {     // [NE,32] f16
  __shared__ __align__(16) _Float16 sm[8][16 * 32 + 16 * 64];
  const int wave = threadIdx.x >> 5;
  const int lane = threadIdx.x & 31;
  _Float16* sA = sm[wave];            // 16x32
  _Float16* sH = sm[wave] + 16 * 32;  // 16x64
  const int n = lane & 15, hi = lane >> 4;
  const int t = blockIdx.x * 8 + wave;
  if (t >= NE / 16) return;

#pragma unroll
  for (int r = 0; r < 16; ++r) {
    if (lane < 8) {
      float4 v = ((const float4*)(ef + (size_t)(t * 16 + r) * 32))[lane];
      v4h h;
      h[0] = (_Float16)v.x; h[1] = (_Float16)v.y;
      h[2] = (_Float16)v.z; h[3] = (_Float16)v.w;
      ((v4h*)(sA + r * 32))[lane] = h;
    }
  }
  v16h a0 = fragA(sA, 0, 32, lane);
#pragma unroll
  for (int j = 0; j < 4; ++j) {
    v8f c = biasC(b1[j * 16 + n]);
    c = wmma_f16(a0, fragBp(w1p, j, lane), c);
#pragma unroll
    for (int r = 0; r < 8; ++r)
      sH[(r + 8 * hi) * 64 + j * 16 + n] = relu16(c[r]);
  }
  v16h h0 = fragA(sH, 0, 64, lane);
  v16h h1 = fragA(sH, 32, 64, lane);
#pragma unroll
  for (int j = 0; j < 2; ++j) {
    v8f c = biasC(b2[j * 16 + n]);
    c = wmma_f16(h0, fragBp(w2p, 0 * 2 + j, lane), c);
    c = wmma_f16(h1, fragBp(w2p, 1 * 2 + j, lane), c);
#pragma unroll
    for (int r = 0; r < 8; ++r)
      outh[(size_t)(t * 16 + r + 8 * hi) * 32 + j * 16 + n] = relu16(c[r]);
  }
}

// ---------------------------------------------------------------------------
// Message + scatter:  m = relu(cat[x[src], ef] @ mw1 + mb1) @ mw2 + mb2
//                     agg[dst] += m           (f32 atomics)
// Grid-stride: weight fragments stay register resident across tiles.
// ---------------------------------------------------------------------------
__global__ void __launch_bounds__(256)
msg_pass_kernel(const _Float16* __restrict__ xh,    // [NN,64] f16
                const _Float16* __restrict__ efh,   // [NE,32] f16
                const long long* __restrict__ ei,   // [2,NE] i64
                const _Float16* __restrict__ mw1p,  // packed [96,64]
                const float* __restrict__ mb1,      // [64]
                const _Float16* __restrict__ mw2p,  // packed [64,64]
                const float* __restrict__ mb2,      // [64]
                float* __restrict__ agg) {          // [NN,64] f32
  __shared__ __align__(16) _Float16 sm[8][16 * 96 + 16 * 64];
  __shared__ int si[8][48];
  const int wave = threadIdx.x >> 5;
  const int lane = threadIdx.x & 31;
  _Float16* sA = sm[wave];            // 16x96 : [x_src | ef]
  _Float16* sH = sm[wave] + 16 * 96;  // 16x64 hidden
  int* ssrc = si[wave];
  int* sdst = si[wave] + 16;
  int* seid = si[wave] + 32;
  const int n = lane & 15, hi = lane >> 4;

  const int ntiles = (2 * NE) / 16;
  for (int t = blockIdx.x * 8 + wave; t < ntiles; t += gridDim.x * 8) {
    // src/dst/eid for the 16 directed edges of this tile
    if (lane < 16) {
      int e2 = t * 16 + lane;
      int s, d, eid;
      if (e2 < NE) { eid = e2;      s = (int)ei[e2];       d = (int)ei[NE + e2]; }
      else         { eid = e2 - NE; s = (int)ei[NE + eid]; d = (int)ei[eid];     }
      ssrc[lane] = s; sdst[lane] = d; seid[lane] = eid;
    }
    // gather A tile: cols 0..63 = x[src] (f16), cols 64..95 = ef[eid] (f16)
#pragma unroll
    for (int r = 0; r < 16; ++r) {
      if (lane < 16)
        ((v4h*)(sA + r * 96))[lane] =
            ((const v4h*)(xh + (size_t)ssrc[r] * 64))[lane];
      else if (lane < 24)
        ((v4h*)(sA + r * 96 + 64))[lane - 16] =
            ((const v4h*)(efh + (size_t)seid[r] * 32))[lane - 16];
    }
    v16h a0 = fragA(sA, 0, 96, lane);
    v16h a1 = fragA(sA, 32, 96, lane);
    v16h a2 = fragA(sA, 64, 96, lane);
    // GEMM1: 96 -> 64 (bias in C), relu
#pragma unroll
    for (int j = 0; j < 4; ++j) {
      v8f c = biasC(mb1[j * 16 + n]);
      c = wmma_f16(a0, fragBp(mw1p, 0 * 4 + j, lane), c);
      c = wmma_f16(a1, fragBp(mw1p, 1 * 4 + j, lane), c);
      c = wmma_f16(a2, fragBp(mw1p, 2 * 4 + j, lane), c);
#pragma unroll
      for (int r = 0; r < 8; ++r)
        sH[(r + 8 * hi) * 64 + j * 16 + n] = relu16(c[r]);
    }
    v16h h0 = fragA(sH, 0, 64, lane);
    v16h h1 = fragA(sH, 32, 64, lane);
    // GEMM2: 64 -> 64 (bias in C, no relu), scatter-add to agg[dst]
#pragma unroll
    for (int j = 0; j < 4; ++j) {
      v8f c = biasC(mb2[j * 16 + n]);
      c = wmma_f16(h0, fragBp(mw2p, 0 * 4 + j, lane), c);
      c = wmma_f16(h1, fragBp(mw2p, 1 * 4 + j, lane), c);
#pragma unroll
      for (int r = 0; r < 8; ++r) {
        int m = r + 8 * hi;
        atomicAdd(&agg[(size_t)sdst[m] * 64 + j * 16 + n], c[r]);
      }
    }
  }
}

// ---------------------------------------------------------------------------
// Node update: x' = relu(cat[agg, x] @ uw + ub)     [NN,64] f16 out
// One 16-row tile per wave.
// ---------------------------------------------------------------------------
__global__ void __launch_bounds__(256)
node_update_kernel(const float* __restrict__ agg,     // [NN,64] f32
                   const _Float16* __restrict__ xh,   // [NN,64] f16
                   const _Float16* __restrict__ uwp,  // packed [128,64]
                   const float* __restrict__ ub,      // [64]
                   _Float16* __restrict__ outh) {     // [NN,64] f16
  __shared__ __align__(16) _Float16 sm[8][16 * 128];
  const int wave = threadIdx.x >> 5;
  const int lane = threadIdx.x & 31;
  _Float16* sA = sm[wave];  // 16x128 : [agg | x]
  const int n = lane & 15, hi = lane >> 4;
  const int t = blockIdx.x * 8 + wave;
  if (t >= NN / 16) return;

#pragma unroll
  for (int r = 0; r < 16; ++r) {
    size_t node = (size_t)(t * 16 + r);
    if (lane < 16) {
      float4 v = ((const float4*)(agg + node * 64))[lane];
      v4h h;
      h[0] = (_Float16)v.x; h[1] = (_Float16)v.y;
      h[2] = (_Float16)v.z; h[3] = (_Float16)v.w;
      ((v4h*)(sA + r * 128))[lane] = h;
    } else {
      ((v4h*)(sA + r * 128 + 64))[lane - 16] =
          ((const v4h*)(xh + node * 64))[lane - 16];
    }
  }
  v16h a0 = fragA(sA, 0, 128, lane);
  v16h a1 = fragA(sA, 32, 128, lane);
  v16h a2 = fragA(sA, 64, 128, lane);
  v16h a3 = fragA(sA, 96, 128, lane);
#pragma unroll
  for (int j = 0; j < 4; ++j) {
    v8f c = biasC(ub[j * 16 + n]);
    c = wmma_f16(a0, fragBp(uwp, 0 * 4 + j, lane), c);
    c = wmma_f16(a1, fragBp(uwp, 1 * 4 + j, lane), c);
    c = wmma_f16(a2, fragBp(uwp, 2 * 4 + j, lane), c);
    c = wmma_f16(a3, fragBp(uwp, 3 * 4 + j, lane), c);
#pragma unroll
    for (int r = 0; r < 8; ++r)
      outh[(size_t)(t * 16 + r + 8 * hi) * 64 + j * 16 + n] = relu16(c[r]);
  }
}

// ---------------------------------------------------------------------------
// Attention-pooling epilogue (f32 scalar; negligible traffic)
// ---------------------------------------------------------------------------
__global__ void gate_kernel(const _Float16* __restrict__ x2h,
                            const float* __restrict__ gw, const float* __restrict__ gb,
                            const long long* __restrict__ batch,
                            float* __restrict__ gate, float* __restrict__ gmax) {
  int i = blockIdx.x * blockDim.x + threadIdx.x;
  if (i >= NN) return;
  float acc = gb[0];
#pragma unroll 8
  for (int d = 0; d < 64; ++d) acc += (float)x2h[(size_t)i * 64 + d] * gw[d];
  gate[i] = acc;
  int g = (int)batch[i];
  // atomic float-max (gmax initialized to -inf)
  if (acc >= 0.f) atomicMax((int*)&gmax[g], __float_as_int(acc));
  else            atomicMin((unsigned int*)&gmax[g], __float_as_uint(acc));
}

__global__ void expsum_kernel(const float* __restrict__ gate,
                              const long long* __restrict__ batch,
                              const float* __restrict__ gmax,
                              float* __restrict__ gexp, float* __restrict__ gsum) {
  int i = blockIdx.x * blockDim.x + threadIdx.x;
  if (i >= NN) return;
  int g = (int)batch[i];
  float e = expf(gate[i] - gmax[g]);
  gexp[i] = e;
  atomicAdd(&gsum[g], e);
}

__global__ void pool_kernel(const _Float16* __restrict__ x2h,
                            const long long* __restrict__ batch,
                            const float* __restrict__ gexp,
                            const float* __restrict__ gsum,
                            float* __restrict__ pooled) {
  int i = blockIdx.x;   // node
  int d = threadIdx.x;  // feature
  int g = (int)batch[i];
  float w = gexp[i] / (gsum[g] + 1e-16f);
  atomicAdd(&pooled[g * 64 + d], w * (float)x2h[(size_t)i * 64 + d]);
}

__global__ void out_kernel(const float* __restrict__ pooled,
                           const float* __restrict__ ow, const float* __restrict__ ob,
                           float* __restrict__ out) {
  int i = blockIdx.x * blockDim.x + threadIdx.x;
  if (i >= NG * 32) return;
  int g = i >> 5, o = i & 31;
  float acc = ob[o];
#pragma unroll 8
  for (int d = 0; d < 64; ++d) acc += pooled[g * 64 + d] * ow[d * 32 + o];
  out[i] = acc;
}

// ---------------------------------------------------------------------------
// Utilities
// ---------------------------------------------------------------------------
__global__ void fill_kernel(float* p, float v, size_t n) {
  size_t i = (size_t)blockIdx.x * blockDim.x + threadIdx.x;
  if (i < n) p[i] = v;
}

// ---------------------------------------------------------------------------
extern "C" void kernel_launch(void* const* d_in, const int* in_sizes, int n_in,
                              void* d_out, int out_size, void* d_ws, size_t ws_size,
                              hipStream_t stream) {
  (void)in_sizes; (void)n_in; (void)out_size; (void)ws_size;
  const float*     node_features = (const float*)d_in[0];
  const long long* edge_index    = (const long long*)d_in[1];
  const float*     edge_features = (const float*)d_in[2];
  const long long* batch         = (const long long*)d_in[3];
  const float* wn1 = (const float*)d_in[4];  const float* bn1 = (const float*)d_in[5];
  const float* wn2 = (const float*)d_in[6];  const float* bn2 = (const float*)d_in[7];
  const float* we1 = (const float*)d_in[8];  const float* be1 = (const float*)d_in[9];
  const float* we2 = (const float*)d_in[10]; const float* be2 = (const float*)d_in[11];
  const float* g1_mw1 = (const float*)d_in[12]; const float* g1_mb1 = (const float*)d_in[13];
  const float* g1_mw2 = (const float*)d_in[14]; const float* g1_mb2 = (const float*)d_in[15];
  const float* g1_uw  = (const float*)d_in[16]; const float* g1_ub  = (const float*)d_in[17];
  const float* g2_mw1 = (const float*)d_in[18]; const float* g2_mb1 = (const float*)d_in[19];
  const float* g2_mw2 = (const float*)d_in[20]; const float* g2_mb2 = (const float*)d_in[21];
  const float* g2_uw  = (const float*)d_in[22]; const float* g2_ub  = (const float*)d_in[23];
  const float* gate_w = (const float*)d_in[24]; const float* gate_b = (const float*)d_in[25];
  const float* out_w  = (const float*)d_in[26]; const float* out_b  = (const float*)d_in[27];

  // carve workspace (~84 MB)
  char* ws = (char*)d_ws;
  size_t off = 0;
  auto carve = [&](size_t bytes) -> char* {
    char* p = ws + off;
    off = (off + bytes + 255) & ~(size_t)255;
    return p;
  };
  _Float16* nfh = (_Float16*)carve((size_t)NN * 64 * 2);
  _Float16* efh = (_Float16*)carve((size_t)NE * 32 * 2);
  _Float16* x1h = (_Float16*)carve((size_t)NN * 64 * 2);
  _Float16* x2h = (_Float16*)carve((size_t)NN * 64 * 2);
  float* agg    = (float*)carve((size_t)NN * 64 * 4);
  float* gate   = (float*)carve((size_t)NN * 4);
  float* gexp   = (float*)carve((size_t)NN * 4);
  float* gmax   = (float*)carve(NG * 4);
  float* gsum   = (float*)carve(NG * 4);
  float* pooled = (float*)carve(NG * 64 * 4);
  _Float16* p_wn1 = (_Float16*)carve(64 * 128 * 2);
  _Float16* p_wn2 = (_Float16*)carve(128 * 64 * 2);
  _Float16* p_we1 = (_Float16*)carve(32 * 64 * 2);
  _Float16* p_we2 = (_Float16*)carve(64 * 32 * 2);
  _Float16* p_g1mw1 = (_Float16*)carve(96 * 64 * 2);
  _Float16* p_g1mw2 = (_Float16*)carve(64 * 64 * 2);
  _Float16* p_g1uw  = (_Float16*)carve(128 * 64 * 2);
  _Float16* p_g2mw1 = (_Float16*)carve(96 * 64 * 2);
  _Float16* p_g2mw2 = (_Float16*)carve(64 * 64 * 2);
  _Float16* p_g2uw  = (_Float16*)carve(128 * 64 * 2);

  auto repack = [&](const float* s, _Float16* d, int K, int Nt) {
    repackB_kernel<<<(K * Nt + 255) / 256, 256, 0, stream>>>(s, d, K, Nt);
  };
  repack(wn1, p_wn1, 64, 128);   repack(wn2, p_wn2, 128, 64);
  repack(we1, p_we1, 32, 64);    repack(we2, p_we2, 64, 32);
  repack(g1_mw1, p_g1mw1, 96, 64); repack(g1_mw2, p_g1mw2, 64, 64);
  repack(g1_uw, p_g1uw, 128, 64);
  repack(g2_mw1, p_g2mw1, 96, 64); repack(g2_mw2, p_g2mw2, 64, 64);
  repack(g2_uw, p_g2uw, 128, 64);

  const int nodeBlocks = (NN / 16 + 7) / 8;  // one 16-row tile per wave
  const int edgeBlocks = (NE / 16 + 7) / 8;
  node_combiner_kernel<<<nodeBlocks, 256, 0, stream>>>(node_features, p_wn1, bn1,
                                                       p_wn2, bn2, nfh);
  edge_combiner_kernel<<<edgeBlocks, 256, 0, stream>>>(edge_features, p_we1, be1,
                                                       p_we2, be2, efh);

  // ---- GNN layer 1 ----
  fill_kernel<<<((size_t)NN * 64 + 255) / 256, 256, 0, stream>>>(agg, 0.f,
                                                                 (size_t)NN * 64);
  msg_pass_kernel<<<4096, 256, 0, stream>>>(nfh, efh, edge_index, p_g1mw1, g1_mb1,
                                            p_g1mw2, g1_mb2, agg);
  node_update_kernel<<<nodeBlocks, 256, 0, stream>>>(agg, nfh, p_g1uw, g1_ub, x1h);

  // ---- GNN layer 2 ----
  fill_kernel<<<((size_t)NN * 64 + 255) / 256, 256, 0, stream>>>(agg, 0.f,
                                                                 (size_t)NN * 64);
  msg_pass_kernel<<<4096, 256, 0, stream>>>(x1h, efh, edge_index, p_g2mw1, g2_mb1,
                                            p_g2mw2, g2_mb2, agg);
  node_update_kernel<<<nodeBlocks, 256, 0, stream>>>(agg, x1h, p_g2uw, g2_ub, x2h);

  // ---- global attention pooling + head ----
  fill_kernel<<<1, 256, 0, stream>>>(gmax, -INFINITY, (size_t)NG);
  fill_kernel<<<1, 256, 0, stream>>>(gsum, 0.f, (size_t)NG);
  fill_kernel<<<(NG * 64 + 255) / 256, 256, 0, stream>>>(pooled, 0.f,
                                                         (size_t)NG * 64);
  gate_kernel<<<(NN + 255) / 256, 256, 0, stream>>>(x2h, gate_w, gate_b, batch,
                                                    gate, gmax);
  expsum_kernel<<<(NN + 255) / 256, 256, 0, stream>>>(gate, batch, gmax, gexp, gsum);
  pool_kernel<<<NN, 64, 0, stream>>>(x2h, batch, gexp, gsum, pooled);
  out_kernel<<<(NG * 32 + 255) / 256, 256, 0, stream>>>(pooled, out_w, out_b,
                                                        (float*)d_out);
}
